// IngridientDecoder_20633022890254
// MI455X (gfx1250) — matrix operand order repaired and checked by reading, not compiled, not run on hardware
//
#include <hip/hip_runtime.h>

// Problem constants
#define B_  1024
#define G_  2048
#define E_  256
#define H_  32
#define V_  14463
#define T_  18
#define S_  19          // T+1
#define GATES_ 128      // 4*H

typedef __bf16 v16bf __attribute__((ext_vector_type(16)));
typedef float  v8f   __attribute__((ext_vector_type(8)));

union BF16Tile { uint4 q[2]; v16bf v; };

__device__ __forceinline__ unsigned short f2bf(float f) {
    unsigned u = __float_as_uint(f);
    u += 0x7FFFu + ((u >> 16) & 1u);     // round-to-nearest-even
    return (unsigned short)(u >> 16);
}

__device__ __forceinline__ float sigmoidf_(float x) {
    return 1.0f / (1.0f + __expf(-x));
}

// ---- WMMA tile loaders (bf16, NT GEMM: A [M,K] row-major, B from W [N,K] row-major) ----
// A 16x32 tile as loaded here (contiguous k, not AMD's table order): lanes 0-15 hold M=lane, K=0..15; lanes 16-31 hold M=lane-16, K=16..31
__device__ __forceinline__ BF16Tile load_a16x32(const unsigned short* __restrict__ base, int ld, int lane) {
    BF16Tile t;
    int row = lane & 15;
    int kb  = (lane >> 4) << 4;                   // 0 or 16
    const unsigned short* p = base + row * ld + kb;
    t.q[0] = *(const uint4*)(p);                  // K kb..kb+7
    t.q[1] = *(const uint4*)(p + 8);              // K kb+8..kb+15
    return t;
}
// B 32x16 tile as loaded here (same contiguous k as A): lane n%16 = column; lanes 0-15 hold K=0..15, lanes 16-31 hold K=16..31
__device__ __forceinline__ BF16Tile load_b32x16(const unsigned short* __restrict__ base, int ld, int lane) {
    BF16Tile t;
    int col = lane & 15;
    int kb  = (lane >> 4) << 4;                   // 0 or 16
    const unsigned short* p = base + col * ld + kb;
    t.q[0] = *(const uint4*)(p);
    t.q[1] = *(const uint4*)(p + 8);
    return t;
}

// ---------------------------------------------------------------------------
// f32 -> bf16 streaming convert
__global__ void k_conv_bf16(const float* __restrict__ src, unsigned short* __restrict__ dst, int n) {
    int i = blockIdx.x * blockDim.x + threadIdx.x;
    if (i < n) dst[i] = f2bf(src[i]);
}

// Embedding gather into inputs_bf16[b, 1+t, :]
__global__ void k_gather(const int* __restrict__ sent, const float* __restrict__ table,
                         unsigned short* __restrict__ inp) {
    int i = blockIdx.x * blockDim.x + threadIdx.x;      // over B*T*E
    if (i >= B_ * T_ * E_) return;
    int e  = i & (E_ - 1);
    int bt = i >> 8;                                     // E_ == 256
    int t  = bt % T_;
    int b  = bt / T_;
    int tok = sent[b * T_ + t];
    inp[b * (S_ * E_) + (t + 1) * E_ + e] = f2bf(table[tok * E_ + e]);
}

// init_embed = graph @ W_g2e^T + b_g2e  ->  inputs_bf16[:, 0, :]
// grid (64, 16), block 32 (one wave per 16x16 tile), K loop 2048/32
__global__ void k_gemm_g2e(const unsigned short* __restrict__ A,   // [B,G] bf16
                           const unsigned short* __restrict__ W,   // [E,G] bf16
                           const float* __restrict__ bias,         // [E]
                           unsigned short* __restrict__ inp) {     // [B,S,E] bf16
    int lane = threadIdx.x & 31;
    int m0 = blockIdx.x << 4;
    int n0 = blockIdx.y << 4;
    int n  = n0 + (lane & 15);
    float bv = bias[n];
    v8f acc;
    #pragma unroll
    for (int i = 0; i < 8; ++i) acc[i] = bv;
    for (int k0 = 0; k0 < G_; k0 += 32) {
        __builtin_prefetch(A + m0 * G_ + k0 + 128, 0, 0);
        BF16Tile a = load_a16x32(A + m0 * G_ + k0, G_, lane);
        BF16Tile b = load_b32x16(W + n0 * G_ + k0, G_, lane);
        acc = __builtin_amdgcn_wmma_f32_16x16x32_bf16(false, a.v, false, b.v,
                                                      (short)0, acc, false, false);
    }
    int mb = m0 + ((lane >> 4) << 3);
    #pragma unroll
    for (int r = 0; r < 8; ++r)
        inp[(mb + r) * (S_ * E_) + n] = f2bf(acc[r]);    // s = 0 slice
}

// gx0 = inputs @ W_ih0^T + (b_ih0 + b_hh0)   M=19456, N=128, K=256
// grid (1216, 8), block 32
__global__ void k_gemm_gx0(const unsigned short* __restrict__ A,   // [B*S, E] bf16
                           const unsigned short* __restrict__ W,   // [4H, E] bf16
                           const float* __restrict__ bih,
                           const float* __restrict__ bhh,
                           float* __restrict__ gx) {               // [B*S, 4H] f32
    int lane = threadIdx.x & 31;
    int m0 = blockIdx.x << 4;
    int n0 = blockIdx.y << 4;
    int n  = n0 + (lane & 15);
    float bv = bih[n] + bhh[n];
    v8f acc;
    #pragma unroll
    for (int i = 0; i < 8; ++i) acc[i] = bv;
    #pragma unroll
    for (int k0 = 0; k0 < E_; k0 += 32) {
        BF16Tile a = load_a16x32(A + m0 * E_ + k0, E_, lane);
        BF16Tile b = load_b32x16(W + n0 * E_ + k0, E_, lane);
        acc = __builtin_amdgcn_wmma_f32_16x16x32_bf16(false, a.v, false, b.v,
                                                      (short)0, acc, false, false);
    }
    int mb = m0 + ((lane >> 4) << 3);
    #pragma unroll
    for (int r = 0; r < 8; ++r)
        gx[(mb + r) * GATES_ + n] = acc[r];
}

// Fused 2-layer LSTM scan. block = 256 threads = 8 batch rows x 32 hidden units.
// Weights LDS-resident; serial over 19 steps; emits h2 (layer-1 h) as bf16 rows [B*T, 32].
__global__ void k_lstm(const float* __restrict__ gx0,      // [B*S, 4H]
                       const float* __restrict__ gWhh0,    // [4H, H]
                       const float* __restrict__ gWih1,    // [4H, H]
                       const float* __restrict__ gWhh1,    // [4H, H]
                       const float* __restrict__ bih1,
                       const float* __restrict__ bhh1,
                       unsigned short* __restrict__ h2) {  // [B*T, H] bf16
    __shared__ float sWhh0[GATES_ * H_];
    __shared__ float sWih1[GATES_ * H_];
    __shared__ float sWhh1[GATES_ * H_];
    __shared__ float sB1[GATES_];
    __shared__ float h0s[8][H_];
    __shared__ float h1s[8][H_];

    int tid = threadIdx.x;
    for (int i = tid; i < GATES_ * H_; i += 256) {
        sWhh0[i] = gWhh0[i];
        sWih1[i] = gWih1[i];
        sWhh1[i] = gWhh1[i];
    }
    if (tid < GATES_) sB1[tid] = bih1[tid] + bhh1[tid];

    int r = tid >> 5;          // local batch row 0..7
    int j = tid & 31;          // hidden unit 0..31
    int b = blockIdx.x * 8 + r;
    h0s[r][j] = 0.0f;
    h1s[r][j] = 0.0f;
    float c0 = 0.0f, c1 = 0.0f;
    __syncthreads();

    for (int t = 0; t < S_; ++t) {
        // ---- layer 0: gates = gx0[b,t,:] + h0 @ Whh0^T ----
        const float* gx = gx0 + (b * S_ + t) * GATES_;
        float g[4];
        #pragma unroll
        for (int k = 0; k < 4; ++k) {
            int u = k * H_ + j;
            float s = gx[u];
            const float* w = &sWhh0[u * H_];
            #pragma unroll
            for (int kk = 0; kk < H_; ++kk) s += h0s[r][kk] * w[kk];
            g[k] = s;
        }
        float ig = sigmoidf_(g[0]), fg = sigmoidf_(g[1]);
        float gg = tanhf(g[2]),     og = sigmoidf_(g[3]);
        c0 = fg * c0 + ig * gg;
        float h0n = og * tanhf(c0);
        __syncthreads();               // old h0 fully consumed
        h0s[r][j] = h0n;
        __syncthreads();               // new h0 visible

        // ---- layer 1: gates = (b_ih1+b_hh1) + h0 @ Wih1^T + h1 @ Whh1^T ----
        #pragma unroll
        for (int k = 0; k < 4; ++k) {
            int u = k * H_ + j;
            float s = sB1[u];
            const float* wa = &sWih1[u * H_];
            const float* wb = &sWhh1[u * H_];
            #pragma unroll
            for (int kk = 0; kk < H_; ++kk)
                s += h0s[r][kk] * wa[kk] + h1s[r][kk] * wb[kk];
            g[k] = s;
        }
        ig = sigmoidf_(g[0]); fg = sigmoidf_(g[1]);
        gg = tanhf(g[2]);     og = sigmoidf_(g[3]);
        c1 = fg * c1 + ig * gg;
        float h1n = og * tanhf(c1);
        __syncthreads();               // old h1 fully consumed
        h1s[r][j] = h1n;
        __syncthreads();

        if (t < T_)                    // feats uses h2 steps 0..T-1
            h2[(b * T_ + t) * H_ + j] = f2bf(h1n);
    }
}

// out = h2 @ W_out^T + b_out   M=18432, N=14463, K=32 -> one WMMA per tile, NT stores.
// grid (1152, 226), block 128 (4 waves; wave w handles N-tile blockIdx.y*4+w)
__global__ void k_gemm_out(const unsigned short* __restrict__ A,   // [B*T, 32] bf16
                           const unsigned short* __restrict__ W,   // [V, 32] bf16
                           const float* __restrict__ bias,         // [V]
                           float* __restrict__ out) {              // [B*T, V]
    int lane = threadIdx.x & 31;
    int wave = threadIdx.x >> 5;
    int m0 = blockIdx.x << 4;
    int n0 = ((blockIdx.y << 2) + wave) << 4;
    int n  = n0 + (lane & 15);
    int nc = (n < V_) ? n : (V_ - 1);              // clamp loads, guard stores
    float bv = (n < V_) ? bias[n] : 0.0f;
    v8f acc;
    #pragma unroll
    for (int i = 0; i < 8; ++i) acc[i] = bv;

    BF16Tile a = load_a16x32(A + m0 * H_, H_, lane);
    BF16Tile b;
    {
        int kb = (lane >> 4) << 4;
        const unsigned short* p = W + nc * H_ + kb;
        b.q[0] = *(const uint4*)(p);
        b.q[1] = *(const uint4*)(p + 8);
    }
    acc = __builtin_amdgcn_wmma_f32_16x16x32_bf16(false, a.v, false, b.v,
                                                  (short)0, acc, false, false);
    int mb = m0 + ((lane >> 4) << 3);
    if (n < V_) {
        #pragma unroll
        for (int r = 0; r < 8; ++r)
            __builtin_nontemporal_store(acc[r], &out[(size_t)(mb + r) * V_ + n]);
    }
}

// ---------------------------------------------------------------------------
extern "C" void kernel_launch(void* const* d_in, const int* in_sizes, int n_in,
                              void* d_out, int out_size, void* d_ws, size_t ws_size,
                              hipStream_t stream) {
    const float* graph = (const float*)d_in[0];
    const int*   sent  = (const int*)  d_in[1];
    const float* Wg2e  = (const float*)d_in[2];
    const float* bg2e  = (const float*)d_in[3];
    const float* emb   = (const float*)d_in[4];
    const float* Wih0  = (const float*)d_in[5];
    const float* Whh0  = (const float*)d_in[6];
    const float* bih0  = (const float*)d_in[7];
    const float* bhh0  = (const float*)d_in[8];
    const float* Wih1  = (const float*)d_in[9];
    const float* Whh1  = (const float*)d_in[10];
    const float* bih1  = (const float*)d_in[11];
    const float* bhh1  = (const float*)d_in[12];
    const float* Wout  = (const float*)d_in[13];
    const float* bout  = (const float*)d_in[14];
    float* out = (float*)d_out;

    char* ws = (char*)d_ws;
    size_t off = 0;
    auto take = [&](size_t bytes) {
        char* p = ws + off;
        off = (off + bytes + 255) & ~(size_t)255;
        return p;
    };
    unsigned short* wsGraph = (unsigned short*)take((size_t)B_ * G_ * 2);       // 4.00 MB
    unsigned short* wsWg    = (unsigned short*)take((size_t)E_ * G_ * 2);       // 1.00 MB
    unsigned short* wsInp   = (unsigned short*)take((size_t)B_ * S_ * E_ * 2);  // 9.50 MB
    unsigned short* wsWih0  = (unsigned short*)take((size_t)GATES_ * E_ * 2);   // 64 KB
    float*          wsGx    = (float*)         take((size_t)B_ * S_ * GATES_ * 4); // 9.50 MB
    unsigned short* wsH2    = (unsigned short*)take((size_t)B_ * T_ * H_ * 2);  // 1.13 MB
    unsigned short* wsWout  = (unsigned short*)take((size_t)V_ * H_ * 2);       // 0.88 MB

    // f32 -> bf16 converts
    k_conv_bf16<<<(B_ * G_ + 255) / 256, 256, 0, stream>>>(graph, wsGraph, B_ * G_);
    k_conv_bf16<<<(E_ * G_ + 255) / 256, 256, 0, stream>>>(Wg2e, wsWg, E_ * G_);
    k_conv_bf16<<<(GATES_ * E_ + 255) / 256, 256, 0, stream>>>(Wih0, wsWih0, GATES_ * E_);
    k_conv_bf16<<<(V_ * H_ + 255) / 256, 256, 0, stream>>>(Wout, wsWout, V_ * H_);

    // caption embeddings -> inputs[:, 1:, :]
    k_gather<<<(B_ * T_ * E_ + 255) / 256, 256, 0, stream>>>(sent, emb, wsInp);

    // init embed -> inputs[:, 0, :]
    k_gemm_g2e<<<dim3(B_ / 16, E_ / 16), 32, 0, stream>>>(wsGraph, wsWg, bg2e, wsInp);

    // layer-0 input gates (bias folded)
    k_gemm_gx0<<<dim3((B_ * S_) / 16, GATES_ / 16), 32, 0, stream>>>(wsInp, wsWih0, bih0, bhh0, wsGx);

    // fused 2-layer recurrence
    k_lstm<<<B_ / 8, 256, 0, stream>>>(wsGx, Whh0, Wih1, Whh1, bih1, bhh1, wsH2);

    // logits (store-bound): N tiles = ceil(14463/16) = 904 = 226 blocks * 4 waves
    k_gemm_out<<<dim3((B_ * T_) / 16, 226), 128, 0, stream>>>(wsH2, wsWout, bout, out);
}
